// CDEModel_78615081386246
// MI455X (gfx1250) — matrix operand order repaired and checked by reading, not compiled, not run on hardware
//
#include <hip/hip_runtime.h>
#include <math.h>

// ---------------------------------------------------------------------------
// Neural CDE forward for MI455X (gfx1250, wave32, WMMA f32_16x16x32_f16).
// B=128, SEG=64, C_IN=32, H=W_MLP=128, C_OUT=1.
// 8 workgroups x 16 batch rows; 256 threads (8 waves) per workgroup.
// Per-WG bottleneck is streaming W3 (1MB f16) from L2 each of 256 stages:
//  - 256KB of W3 is pinned in LDS (4 of each wave's 16 hh rows)
//  - the remaining 12 hh rows are streamed with register ping-pong buffering
// ---------------------------------------------------------------------------

typedef __attribute__((ext_vector_type(16))) _Float16 v16h;
typedef __attribute__((ext_vector_type(8)))  float    v8f;

union F16x16 {
    v16h  h;
    uint4 q[2];
};

// ---------------------------------------------------------------------------
// Weight pre-swizzle: f32 row-major W[K][N]  ->  f16 WMMA B-fragments.
// Fragment (16x16x32 tile): element e (0..15) of lane l holds
//   N = nt*16 + (l & 15),   K = kt*32 + (l>>4)*16 + e
// stored as [nt][kt][lane][e] so each lane loads 2 x 16 bytes.
// ---------------------------------------------------------------------------
__global__ void swizzle_weights(const float* __restrict__ W,
                                _Float16* __restrict__ F, int K, int N) {
    const int KT = K >> 5;
    const int total = K * N;
    for (int i = blockIdx.x * blockDim.x + threadIdx.x; i < total;
         i += gridDim.x * blockDim.x) {
        int e    = i & 15;
        int lane = (i >> 4) & 31;
        int f    = i >> 9;          // fragment index = nt*KT + kt
        int kt   = f % KT;
        int nt   = f / KT;
        int n = nt * 16 + (lane & 15);
        int k = kt * 32 + (lane >> 4) * 16 + e;
        F[i] = (_Float16)W[k * N + n];
    }
}

// B fragment: 2 x 16-byte global loads per lane (pre-swizzled array).
__device__ __forceinline__ F16x16 loadB(const _Float16* __restrict__ Wf,
                                        int KT, int nt, int kt, int lane) {
    F16x16 r;
    const uint4* p = (const uint4*)(Wf + ((size_t)((nt * KT + kt) * 32 + lane) << 4));
    r.q[0] = p[0];
    r.q[1] = p[1];
    return r;
}

// B fragment from the LDS-resident W3 cache.
__device__ __forceinline__ F16x16 loadBc(const uint4* cache, int wave, int hl,
                                         int half, int kt, int lane) {
    F16x16 r;
    const uint4* p = cache + (size_t)(((((wave * 4 + hl) * 2 + half) * 4 + kt) * 32 + lane)) * 2;
    r.q[0] = p[0];
    r.q[1] = p[1];
    return r;
}

// A fragment from LDS row-major f16 [16][128].
// Order used here (matches B swizzle): lane l row M=l&15; elements 0..15 = K0..K0+15
// with K0 = kt*32 + (l>>4)*16  -> two contiguous 16-byte ds reads.
__device__ __forceinline__ F16x16 loadA(const _Float16* Asrc, int kt, int lane) {
    F16x16 r;
    const int m  = lane & 15;
    const int K0 = kt * 32 + ((lane >> 4) << 4);
    const _Float16* row = Asrc + m * 128 + K0;
    r.q[0] = *(const uint4*)(row);
    r.q[1] = *(const uint4*)(row + 8);
    return r;
}

// One MLP layer: Dst(16x128,f16) = relu(A(16x128,f16) @ W(128x128) + bias).
// Wave w owns N-tile w. 4 WMMAs over K.
__device__ __forceinline__ void mlp_layer(const _Float16* Asrc,
                                          const _Float16* __restrict__ Wf,
                                          const float* __restrict__ bias,
                                          _Float16* Dst, int wave, int lane) {
    F16x16 a0 = loadA(Asrc, 0, lane);
    F16x16 a1 = loadA(Asrc, 1, lane);
    F16x16 a2 = loadA(Asrc, 2, lane);
    F16x16 a3 = loadA(Asrc, 3, lane);
    v8f acc = {};
    F16x16 b;
    b = loadB(Wf, 4, wave, 0, lane);
    acc = __builtin_amdgcn_wmma_f32_16x16x32_f16(false, a0.h, false, b.h, (short)0, acc, false, false);
    b = loadB(Wf, 4, wave, 1, lane);
    acc = __builtin_amdgcn_wmma_f32_16x16x32_f16(false, a1.h, false, b.h, (short)0, acc, false, false);
    b = loadB(Wf, 4, wave, 2, lane);
    acc = __builtin_amdgcn_wmma_f32_16x16x32_f16(false, a2.h, false, b.h, (short)0, acc, false, false);
    b = loadB(Wf, 4, wave, 3, lane);
    acc = __builtin_amdgcn_wmma_f32_16x16x32_f16(false, a3.h, false, b.h, (short)0, acc, false, false);

    const float bn  = bias[wave * 16 + (lane & 15)];
    const int   mh  = (lane >> 4) * 8;
    const int   col = wave * 16 + (lane & 15);
#pragma unroll
    for (int r = 0; r < 8; r++) {
        float v = acc[r] + bn;
        v = v > 0.f ? v : 0.f;
        Dst[(mh + r) * 128 + col] = (_Float16)v;
    }
}

// Load the 8 streamed W3 fragments (2 c-halves x 4 K-tiles) for one hh row.
__device__ __forceinline__ void w3_loadfrags(F16x16 b[8],
                                             const _Float16* __restrict__ W3f,
                                             int hh, int lane) {
#pragma unroll
    for (int half = 0; half < 2; half++)
#pragma unroll
        for (int kt = 0; kt < 4; kt++)
            b[half * 4 + kt] = loadB(W3f, 4, hh * 2 + half, kt, lane);
}

// 8 WMMAs + bias + tanh + contraction with xd + cross-lane reduce for one hh.
// g[m][hh] = sum_c tanh((h2 @ W3 + b3))[m][hh*32+c] * xd[m][c]
__device__ __forceinline__ void w3_compute_hh(const F16x16 a[4], const F16x16 b[8],
                                              const float* __restrict__ b3,
                                              const float* xd, float* gdst,
                                              int hh, int lane) {
    const int mh = (lane >> 4) * 8;   // row base for this lane half
    const int cl = lane & 15;         // column-within-tile for this lane
    float part[8];
#pragma unroll
    for (int r = 0; r < 8; r++) part[r] = 0.f;

#pragma unroll
    for (int half = 0; half < 2; half++) {
        const int nt = hh * 2 + half;
        v8f acc = {};
#pragma unroll
        for (int kt = 0; kt < 4; kt++) {
            acc = __builtin_amdgcn_wmma_f32_16x16x32_f16(
                false, a[kt].h, false, b[half * 4 + kt].h, (short)0, acc, false, false);
        }
        const float bn = b3[nt * 16 + cl];
        const int   c  = half * 16 + cl;       // C_IN index held by this lane
#pragma unroll
        for (int r = 0; r < 8; r++) {
            float h3 = tanhf(acc[r] + bn);
            part[r] += h3 * xd[(mh + r) * 32 + c];
        }
    }
    // Reduce across the 16 lanes of each half (halves carry different rows m).
#pragma unroll
    for (int off = 8; off >= 1; off >>= 1) {
#pragma unroll
        for (int r = 0; r < 8; r++)
            part[r] += __shfl_xor(part[r], off, 16);
    }
    if (cl == 0) {
#pragma unroll
        for (int r = 0; r < 8; r++) gdst[(mh + r) * 128 + hh] = part[r];
    }
}

// ---------------------------------------------------------------------------
// Main persistent kernel: one workgroup = 16 batch rows, full 64-step RK4.
// ---------------------------------------------------------------------------
__global__ __launch_bounds__(256) void cde_main(
    const float* __restrict__ coeffs,
    const float* __restrict__ b1, const float* __restrict__ b2,
    const float* __restrict__ b3,
    const float* __restrict__ Wi, const float* __restrict__ bi,
    const float* __restrict__ Wr, const float* __restrict__ br,
    const _Float16* __restrict__ W1f, const _Float16* __restrict__ W2f,
    const _Float16* __restrict__ W3f,
    float* __restrict__ out) {
    __shared__ float    zbuf[2048];     // z      (16 x 128) f32
    __shared__ float    kacc[2048];     // RK4 accumulator
    __shared__ float    gbuf[2048];     // current k = g(t, y)
    __shared__ _Float16 ain[2048];      // stage input y, f16
    __shared__ _Float16 h1b[2048];      // h1, f16
    __shared__ _Float16 h2b[2048];      // h2, f16
    __shared__ float    xds[3][512];    // X'(t) at f=0, f=0.5, t=step+1
    __shared__ uint4    w3cache[16384]; // 256KB: 4 resident hh rows per wave

    const int tid  = threadIdx.x;
    const int lane = tid & 31;
    const int wave = tid >> 5;
    const int wg   = blockIdx.x;        // 0..7 -> batch rows wg*16 .. +15

    // ---- Fill the resident W3 cache via CDNA5 async global->LDS DMA ----
    // cache uint4 index: q | lane(5) | kt(2) | half(1) | hl(2) | wave(3)
    for (int ci = tid; ci < 16384; ci += 256) {
        const int q    = ci & 1;
        const int ln   = (ci >> 1) & 31;
        const int kt   = (ci >> 6) & 3;
        const int half = (ci >> 8) & 1;
        const int hl   = (ci >> 9) & 3;
        const int w    = ci >> 11;
        const int nt   = (w * 16 + hl) * 2 + half;
        const unsigned long long ga =
            (unsigned long long)(W3f + ((size_t)((nt * 4 + kt) * 32 + ln) << 4)) +
            (unsigned long long)q * 16u;
        const unsigned lofs = (unsigned)(uintptr_t)(&w3cache[ci]);
        asm volatile("global_load_async_to_lds_b128 %0, %1, off"
                     :: "v"(lofs), "v"(ga) : "memory");
    }
    asm volatile("s_wait_asynccnt 0x0" ::: "memory");

    // ---- z0 = X(0) @ Wi + bi ;  X(0) = ca[:,0] ----
    for (int o = tid; o < 2048; o += 256) {
        const int m = o >> 7, h = o & 127;
        const int bidx = wg * 16 + m;
        const float* x0 = coeffs + (size_t)bidx * 64 * 128;  // seg 0, ca = [0,32)
        float s = bi[h];
#pragma unroll 8
        for (int c = 0; c < 32; c++) s += x0[c] * Wi[c * 128 + h];
        zbuf[o] = s;
    }
    __syncthreads();

    for (int step = 0; step < 64; step++) {
        // ---- spline derivatives + stage-0 input (y = z) ----
        const int   idx2 = (step + 1 < 64) ? step + 1 : 63;
        const float f2   = (step + 1 < 64) ? 0.f : 1.f;
        for (int o = tid; o < 512; o += 256) {
            const int m = o >> 5, c = o & 31;
            const int bidx = wg * 16 + m;
            const float* seg = coeffs + ((size_t)bidx * 64 + step) * 128;
            const float cbv = seg[32 + c], c2v = seg[64 + c], c3v = seg[96 + c];
            xds[0][o] = cbv;                                  // f = 0
            xds[1][o] = cbv + (c2v + c3v * 0.5f) * 0.5f;      // f = 0.5
            const float* sg2 = coeffs + ((size_t)bidx * 64 + idx2) * 128;
            const float cb2 = sg2[32 + c], c22 = sg2[64 + c], c32 = sg2[96 + c];
            xds[2][o] = cb2 + (c22 + c32 * f2) * f2;          // t = step+1
        }
        for (int o = tid; o < 2048; o += 256) ain[o] = (_Float16)zbuf[o];
        __syncthreads();

        // ---- 4 RK4 stages ----
        for (int st = 0; st < 4; st++) {
            const int xsel = (st == 0) ? 0 : ((st == 3) ? 2 : 1);

            mlp_layer(ain, W1f, b1, h1b, wave, lane);
            __syncthreads();
            mlp_layer(h1b, W2f, b2, h2b, wave, lane);
            __syncthreads();

            // ---- W3 + tanh + contraction ----
            {
                F16x16 a[4];
#pragma unroll
                for (int kt = 0; kt < 4; kt++) a[kt] = loadA(h2b, kt, lane);
                const float* xd = &xds[xsel][0];

                // resident rows (hh local 0..3) from LDS
#pragma unroll
                for (int hl = 0; hl < 4; hl++) {
                    F16x16 b[8];
#pragma unroll
                    for (int half = 0; half < 2; half++)
#pragma unroll
                        for (int kt = 0; kt < 4; kt++)
                            b[half * 4 + kt] = loadBc(w3cache, wave, hl, half, kt, lane);
                    w3_compute_hh(a, b, b3, xd, gbuf, wave * 16 + hl, lane);
                }
                // streamed rows (hh local 4..15), register ping-pong pipeline
                F16x16 bA[8], bB[8];
                w3_loadfrags(bA, W3f, wave * 16 + 4, lane);
                for (int hl = 4; hl < 16; hl += 2) {
                    w3_loadfrags(bB, W3f, wave * 16 + hl + 1, lane);
                    w3_compute_hh(a, bA, b3, xd, gbuf, wave * 16 + hl, lane);
                    if (hl + 2 < 16)
                        w3_loadfrags(bA, W3f, wave * 16 + hl + 2, lane);
                    w3_compute_hh(a, bB, b3, xd, gbuf, wave * 16 + hl + 1, lane);
                }
            }
            __syncthreads();

            // ---- RK4 bookkeeping merged with next stage's input prep ----
            if (st < 3) {
                const float wc = (st == 0) ? 1.f : 2.f;   // kacc weight of this k
                const float bc = (st == 2) ? 1.f : 0.5f;  // y = z + bc * k
                for (int o = tid; o < 2048; o += 256) {
                    const float gv = gbuf[o];
                    kacc[o] = (st == 0) ? gv : kacc[o] + wc * gv;
                    ain[o]  = (_Float16)(zbuf[o] + bc * gv);
                }
            } else {
                for (int o = tid; o < 2048; o += 256)
                    zbuf[o] += (kacc[o] + gbuf[o]) * (1.f / 6.f);
            }
            __syncthreads();
        }
    }

    // ---- readout: sigmoid(z @ Wr + br), C_OUT = 1 ----
    if (tid < 16) {
        float s = br[0];
#pragma unroll 8
        for (int h = 0; h < 128; h++) s += zbuf[tid * 128 + h] * Wr[h];
        out[wg * 16 + tid] = 1.f / (1.f + expf(-s));
    }
}

// ---------------------------------------------------------------------------
extern "C" void kernel_launch(void* const* d_in, const int* in_sizes, int n_in,
                              void* d_out, int out_size, void* d_ws, size_t ws_size,
                              hipStream_t stream) {
    (void)in_sizes; (void)n_in; (void)out_size; (void)ws_size;
    const float* coeffs = (const float*)d_in[0];
    const float* W1 = (const float*)d_in[1];
    const float* b1 = (const float*)d_in[2];
    const float* W2 = (const float*)d_in[3];
    const float* b2 = (const float*)d_in[4];
    const float* W3 = (const float*)d_in[5];
    const float* b3 = (const float*)d_in[6];
    const float* Wi = (const float*)d_in[7];
    const float* bi = (const float*)d_in[8];
    const float* Wr = (const float*)d_in[9];
    const float* br = (const float*)d_in[10];

    _Float16* W1f = (_Float16*)d_ws;           // 128*128
    _Float16* W2f = W1f + 128 * 128;           // 128*128
    _Float16* W3f = W2f + 128 * 128;           // 128*4096

    swizzle_weights<<<dim3(64), dim3(256), 0, stream>>>(W1, W1f, 128, 128);
    swizzle_weights<<<dim3(64), dim3(256), 0, stream>>>(W2, W2f, 128, 128);
    swizzle_weights<<<dim3(512), dim3(256), 0, stream>>>(W3, W3f, 128, 4096);

    cde_main<<<dim3(8), dim3(256), 0, stream>>>(
        coeffs, b1, b2, b3, Wi, bi, Wr, br, W1f, W2f, W3f, (float*)d_out);
}
